// SelfAttention_52939766891092
// MI455X (gfx1250) — matrix operand order repaired and checked by reading, not compiled, not run on hardware
//
#include <hip/hip_runtime.h>
#include <cstdint>
#include <cstddef>

// ---------------------------------------------------------------------------
// CDNA5 / gfx1250 self-attention (GQA + RoPE + causal flash attention)
// - all matmuls on v_wmma_f32_16x16x32_bf16 (f32 accumulate)
// - LDS tile staging via GLOBAL_LOAD_ASYNC_TO_LDS_B128 (ASYNCcnt-tracked)
// ---------------------------------------------------------------------------

typedef __bf16 bf16;
typedef __attribute__((ext_vector_type(16))) __bf16 v16bf;
typedef __attribute__((ext_vector_type(8)))  __bf16 v8bf;
typedef __attribute__((ext_vector_type(8)))  float  v8f;

#define B_    2
#define L_    2048
#define DIM_  2048
#define NH_   16
#define NKV_  4
#define HD_   128
#define NREP_ 4
#define SCALE_ 0.08838834764831845f   // 1/sqrt(128), folded into Q at RoPE time

#if defined(__AMDGCN__) &&                                          \
    __has_builtin(__builtin_amdgcn_global_load_async_to_lds_b128) && \
    __has_builtin(__builtin_amdgcn_s_wait_asynccnt)
#define USE_ASYNC 1
#else
#define USE_ASYNC 0
#endif

#if USE_ASYNC
typedef int v4i_ __attribute__((vector_size(16)));
typedef __attribute__((address_space(1))) v4i_ as1_v4i;
typedef __attribute__((address_space(3))) v4i_ as3_v4i;
#define ASYNC_WAIT(n) __builtin_amdgcn_s_wait_asynccnt(n)
#else
#define ASYNC_WAIT(n)
#endif

// Copy 16B global -> LDS. Async (ASYNCcnt, no VGPR round trip) when available.
__device__ __forceinline__ void cp_b128(const bf16* __restrict__ g, bf16* l) {
#if USE_ASYNC
  __builtin_amdgcn_global_load_async_to_lds_b128((as1_v4i*)g, (as3_v4i*)l, 0, 0);
#else
  *(uint4*)l = *(const uint4*)g;
#endif
}

// Load a 16-bf16 fragment as two 16B chunks.
__device__ __forceinline__ v16bf frag2(const bf16* p0, const bf16* p1) {
  union { uint4 u[2]; v16bf v; } f;
  f.u[0] = *(const uint4*)p0;
  f.u[1] = *(const uint4*)p1;
  return f.v;
}

__device__ __forceinline__ v8f wmma_bf16f32(v16bf a, v16bf b, v8f c) {
  return __builtin_amdgcn_wmma_f32_16x16x32_bf16(false, a, false, b, (short)0, c,
                                                 false, false);
}

// ---------------------------------------------------------------------------
// f32 -> bf16 cast
// ---------------------------------------------------------------------------
__global__ __launch_bounds__(256) void cast_kernel(const float* __restrict__ s,
                                                   bf16* __restrict__ d, int n) {
  int i = blockIdx.x * 256 + threadIdx.x;
  if (i < n) d[i] = (bf16)s[i];
}

// ---------------------------------------------------------------------------
// C[M,N] = A[M,K](bf16,row-major) * W[N,K](bf16,row-major)^T
// Block tile 128x128, K-step 32, double-buffered async LDS staging.
// 8 waves; each wave computes 2(M) x 4(N) WMMA tiles (8 WMMAs / K-step).
// Output format (f32 vs bf16) is a compile-time template parameter.
// ---------------------------------------------------------------------------
template <bool BF16OUT>
__global__ __launch_bounds__(256) void gemm_bf16_kernel(
    const bf16* __restrict__ A, const bf16* __restrict__ W,
    void* __restrict__ Cout, int M, int N, int K) {
  __shared__ alignas(16) bf16 As[2][128 * 32];
  __shared__ alignas(16) bf16 Bs[2][128 * 32];

  const int tid  = threadIdx.x;
  const int wid  = tid >> 5;
  const int lane = tid & 31;
  const int ln   = lane & 15;
  const int hf   = lane >> 4;
  const int bm   = blockIdx.y * 128;
  const int bn   = blockIdx.x * 128;
  const int wm   = (wid & 3) * 32;   // 4 waves along M (2 tiles each)
  const int wn   = (wid >> 2) * 64;  // 2 waves along N (4 tiles each)

  // Stage one 128x32 A tile + 128x32 B tile into buffer `buf`.
  auto stage = [&](int buf, int k0) {
#pragma unroll
    for (int i = 0; i < 2; ++i) {
      int idx = tid * 2 + i;             // 0..511
      int row = idx >> 2, cg = idx & 3;  // 4 groups of 8 bf16 per row
      cp_b128(&A[(size_t)(bm + row) * K + k0 + cg * 8],
              &As[buf][row * 32 + cg * 8]);
      cp_b128(&W[(size_t)(bn + row) * K + k0 + cg * 8],
              &Bs[buf][row * 32 + cg * 8]);
    }
  };

  v8f acc[2][4];
  for (int tm = 0; tm < 2; ++tm)
    for (int tn = 0; tn < 4; ++tn)
      for (int r = 0; r < 8; ++r) acc[tm][tn][r] = 0.f;

  stage(0, 0);
  const int steps = K / 32;
  for (int it = 0; it < steps; ++it) {
    const int cur = it & 1;
    if (it + 1 < steps) {
      stage(cur ^ 1, (it + 1) * 32);
      ASYNC_WAIT(4);   // in-order retirement: current buffer's 4 loads landed
    } else {
      ASYNC_WAIT(0);
    }
    __syncthreads();   // staged data visible to all waves

    v16bf af[2], bfr[4];
#pragma unroll
    for (int t = 0; t < 2; ++t) {
      const bf16* ap = &As[cur][(wm + t * 16 + ln) * 32];
      af[t] = frag2(ap + 8 * hf, ap + 16 + 8 * hf);
    }
#pragma unroll
    for (int t = 0; t < 4; ++t) {
      const bf16* bp = &Bs[cur][(wn + t * 16 + ln) * 32];
      bfr[t] = frag2(bp + 8 * hf, bp + 16 + 8 * hf);
    }
#pragma unroll
    for (int tm = 0; tm < 2; ++tm)
#pragma unroll
      for (int tn = 0; tn < 4; ++tn)
        acc[tm][tn] = wmma_bf16f32(af[tm], bfr[tn], acc[tm][tn]);

    __syncthreads();   // all reads of buf `cur` done before it is re-staged
  }

#pragma unroll
  for (int tm = 0; tm < 2; ++tm)
    for (int tn = 0; tn < 4; ++tn)
      for (int r = 0; r < 8; ++r) {
        int row = bm + wm + tm * 16 + r + 8 * hf;
        int col = bn + wn + tn * 16 + ln;
        size_t idx = (size_t)row * N + col;
        if constexpr (BF16OUT)
          ((bf16*)Cout)[idx] = (bf16)acc[tm][tn][r];
        else
          ((float*)Cout)[idx] = acc[tm][tn][r];
      }
}

// ---------------------------------------------------------------------------
// RoPE on Q (bf16 in, scaled bf16 out). One thread per (token, head, d-pair).
// ---------------------------------------------------------------------------
__global__ __launch_bounds__(256) void rope_q_kernel(
    const bf16* __restrict__ qpre, const float* __restrict__ cosT,
    const float* __restrict__ sinT, bf16* __restrict__ qout) {
  int i    = blockIdx.x * 256 + threadIdx.x; // B*L*NH*64
  int dp   = i & 63;
  int rest = i >> 6;
  int h    = rest & (NH_ - 1);
  int bl   = rest >> 4;                      // b*L + l
  int l    = bl & (L_ - 1);
  float c = cosT[l * 64 + dp], s = sinT[l * 64 + dp];
  size_t base = ((size_t)bl * NH_ + h) * HD_ + 2 * dp;
  float t0 = (float)qpre[base], t1 = (float)qpre[base + 1];
  qout[base]     = (bf16)((t0 * c - t1 * s) * SCALE_);
  qout[base + 1] = (bf16)((t0 * s + t1 * c) * SCALE_);
}

// ---------------------------------------------------------------------------
// RoPE on K (f32 in) -> compact bf16 K-cache + expanded f32 keys output.
// ---------------------------------------------------------------------------
__global__ __launch_bounds__(256) void rope_k_kernel(
    const float* __restrict__ kf32, const float* __restrict__ cosT,
    const float* __restrict__ sinT, bf16* __restrict__ kb,
    float* __restrict__ keys) {
  int i    = blockIdx.x * 256 + threadIdx.x; // B*L*NKV*64
  int dp   = i & 63;
  int rest = i >> 6;
  int g    = rest & (NKV_ - 1);
  int bl   = rest >> 2;
  int l    = bl & (L_ - 1);
  int b    = bl >> 11;                       // L_ == 2048
  float c = cosT[l * 64 + dp], s = sinT[l * 64 + dp];
  size_t base = ((size_t)bl * NKV_ + g) * HD_ + 2 * dp;
  float t0 = kf32[base], t1 = kf32[base + 1];
  float r0 = t0 * c - t1 * s, r1 = t0 * s + t1 * c;
  kb[base]     = (bf16)r0;
  kb[base + 1] = (bf16)r1;
#pragma unroll
  for (int j = 0; j < NREP_; ++j) {
    int h = g * NREP_ + j;
    size_t o = (((size_t)b * NH_ + h) * L_ + l) * HD_ + 2 * dp;
    keys[o] = r0;
    keys[o + 1] = r1;
  }
}

// ---------------------------------------------------------------------------
// V: f32 -> compact bf16 V-cache + expanded f32 values output.
// ---------------------------------------------------------------------------
__global__ __launch_bounds__(256) void expand_v_kernel(
    const float* __restrict__ vf32, bf16* __restrict__ vb,
    float* __restrict__ values) {
  int i    = blockIdx.x * 256 + threadIdx.x; // B*L*NKV*HD
  int d    = i & 127;
  int rest = i >> 7;
  int g    = rest & (NKV_ - 1);
  int bl   = rest >> 2;
  int l    = bl & (L_ - 1);
  int b    = bl >> 11;
  float val = vf32[i];
  vb[i] = (bf16)val;
#pragma unroll
  for (int j = 0; j < NREP_; ++j) {
    int h = g * NREP_ + j;
    values[(((size_t)b * NH_ + h) * L_ + l) * HD_ + d] = val;
  }
}

// ---------------------------------------------------------------------------
// Causal flash attention. One workgroup per (b*h, 128 q-rows); 8 waves,
// each wave owns 16 q-rows. K staged async into LDS; V^T staged via VGPRs.
// ---------------------------------------------------------------------------
__global__ __launch_bounds__(256) void attn_kernel(
    const bf16* __restrict__ q, const bf16* __restrict__ k,
    const bf16* __restrict__ v, bf16* __restrict__ attn) {
  __shared__ alignas(16) bf16 Ks[32 * 128];   // [key][d]
  __shared__ alignas(16) bf16 VTs[128 * 32];  // [d][key]
  __shared__ alignas(16) bf16 Ps[8][16 * 32]; // per-wave P staging

  const int bh  = blockIdx.x;
  const int b   = bh / NH_;
  const int h   = bh % NH_;
  const int g   = h / NREP_;
  const int qt  = blockIdx.y;
  const int tid = threadIdx.x;
  const int wid = tid >> 5;
  const int lane = tid & 31;
  const int ln  = lane & 15;
  const int hf  = lane >> 4;
  const int q0  = qt * 128 + wid * 16;

  // Q fragments (A-operand), 16 rows x 128 d, scale already folded in.
  v16bf qf[4];
  {
    const bf16* qrow = q + ((size_t)(b * L_ + q0 + ln) * NH_ + h) * HD_;
#pragma unroll
    for (int kf = 0; kf < 4; ++kf)
      qf[kf] = frag2(qrow + kf * 32 + 8 * hf, qrow + kf * 32 + 16 + 8 * hf);
  }

  float m_i[8], l_i[8];
  v8f o[8];
  for (int r = 0; r < 8; ++r) { m_i[r] = -1e30f; l_i[r] = 0.f; }
  for (int j = 0; j < 8; ++j)
    for (int r = 0; r < 8; ++r) o[j][r] = 0.f;

  const int kb_end  = (qt + 1) * 4;        // 32-key blocks (uniform per block)
  const int my_last = (q0 + 15) >> 5;      // last block this wave needs

  for (int kb = 0; kb < kb_end; ++kb) {
    __syncthreads();
    // Cooperative K/V block staging: 32 keys x 128 d
#pragma unroll
    for (int i = 0; i < 2; ++i) {
      int idx = tid * 2 + i;
      int row = idx >> 4;              // key 0..31
      int d   = (idx & 15) * 8;        // d group of 8 bf16
      size_t gidx = ((size_t)(b * L_ + kb * 32 + row) * NKV_ + g) * HD_ + d;
      cp_b128(&k[gidx], &Ks[row * 128 + d]);          // async when available
      v8bf vv = *(const v8bf*)&v[gidx];               // V needs transpose
#pragma unroll
      for (int e = 0; e < 8; ++e) VTs[(d + e) * 32 + row] = vv[e];
    }
    if (kb + 1 < kb_end) {
      __builtin_prefetch(
          &k[((size_t)(b * L_ + (kb + 1) * 32 + (tid >> 3)) * NKV_ + g) * HD_ +
             (tid & 7) * 16], 0, 0);
    }
    ASYNC_WAIT(0);
    __syncthreads();
    if (kb > my_last) continue;  // no barriers below -> safe wave-level skip

    // S = Q * K^T : two 16x16 tiles (keys kb*32+ln, kb*32+16+ln)
    v8f s0, s1;
    for (int r = 0; r < 8; ++r) { s0[r] = 0.f; s1[r] = 0.f; }
#pragma unroll
    for (int kf = 0; kf < 4; ++kf) {
      const bf16* kp0 = &Ks[ln * 128 + kf * 32];
      const bf16* kp1 = &Ks[(16 + ln) * 128 + kf * 32];
      v16bf b0 = frag2(kp0 + 8 * hf, kp0 + 16 + 8 * hf);
      v16bf b1 = frag2(kp1 + 8 * hf, kp1 + 16 + 8 * hf);
      s0 = wmma_bf16f32(qf[kf], b0, s0);
      s1 = wmma_bf16f32(qf[kf], b1, s1);
    }

    // Online softmax (rows live across the 16 lanes of each half)
    const int kbase = kb * 32;
#pragma unroll
    for (int r = 0; r < 8; ++r) {
      const int qrow = q0 + r + 8 * hf;
      if (kbase + ln > qrow)      s0[r] = -1e30f;
      if (kbase + 16 + ln > qrow) s1[r] = -1e30f;
      float mx = fmaxf(s0[r], s1[r]);
      for (int off = 8; off > 0; off >>= 1)
        mx = fmaxf(mx, __shfl_xor(mx, off, 16));
      const float mn    = fmaxf(m_i[r], mx);
      const float alpha = __expf(m_i[r] - mn);
      m_i[r] = mn;
      const float p0 = __expf(s0[r] - mn);
      const float p1 = __expf(s1[r] - mn);
      s0[r] = p0;
      s1[r] = p1;
      float rs = p0 + p1;
      for (int off = 8; off > 0; off >>= 1)
        rs += __shfl_xor(rs, off, 16);
      l_i[r] = l_i[r] * alpha + rs;
#pragma unroll
      for (int j = 0; j < 8; ++j) o[j][r] *= alpha;
    }

    // C-layout P -> A-fragment layout via wave-local LDS round trip
    bf16* ps = &Ps[wid][0];
#pragma unroll
    for (int r = 0; r < 8; ++r) {
      ps[(r + 8 * hf) * 32 + ln]      = (bf16)s0[r];
      ps[(r + 8 * hf) * 32 + 16 + ln] = (bf16)s1[r];
    }
    v16bf pf = frag2(ps + ln * 32 + 8 * hf, ps + ln * 32 + 16 + 8 * hf);

    // O += P * V  (8 N-tiles of 16)
#pragma unroll
    for (int j = 0; j < 8; ++j) {
      const bf16* vp = &VTs[(j * 16 + ln) * 32];
      v16bf vf = frag2(vp + 8 * hf, vp + 16 + 8 * hf);
      o[j] = wmma_bf16f32(pf, vf, o[j]);
    }
  }

  // Normalize and write attention output in (b, l, h*HD) layout (bf16)
#pragma unroll
  for (int r = 0; r < 8; ++r) {
    const float invl = 1.f / l_i[r];
    const int row = q0 + r + 8 * hf;
    bf16* arow = attn + (size_t)(b * L_ + row) * DIM_ + h * HD_;
#pragma unroll
    for (int j = 0; j < 8; ++j)
      arow[j * 16 + ln] = (bf16)(o[j][r] * invl);
  }
}

// ---------------------------------------------------------------------------
// Host launcher
// ---------------------------------------------------------------------------
extern "C" void kernel_launch(void* const* d_in, const int* in_sizes, int n_in,
                              void* d_out, int out_size, void* d_ws,
                              size_t ws_size, hipStream_t stream) {
  (void)in_sizes; (void)n_in; (void)out_size; (void)ws_size;

  const float* x  = (const float*)d_in[0];
  const float* fc = (const float*)d_in[1];
  const float* fs = (const float*)d_in[2];
  const float* wq = (const float*)d_in[3];
  const float* wk = (const float*)d_in[4];
  const float* wv = (const float*)d_in[5];
  const float* wo = (const float*)d_in[6];

  float* out    = (float*)d_out;
  float* keys   = out + (size_t)B_ * L_ * DIM_;
  float* values = keys + (size_t)B_ * NH_ * L_ * HD_;

  // Workspace layout (~113 MB total, 256B-aligned slices)
  char* ws = (char*)d_ws;
  size_t off = 0;
  auto take = [&](size_t bytes) {
    void* p = ws + off;
    off += (bytes + 255) & ~(size_t)255;
    return p;
  };
  const size_t T = (size_t)B_ * L_;  // 4096 tokens
  bf16*  xb    = (bf16*)take(T * DIM_ * 2);
  bf16*  wqb   = (bf16*)take((size_t)DIM_ * DIM_ * 2);
  bf16*  wkb   = (bf16*)take((size_t)(NKV_ * HD_) * DIM_ * 2);
  bf16*  wvb   = (bf16*)take((size_t)(NKV_ * HD_) * DIM_ * 2);
  bf16*  wob   = (bf16*)take((size_t)DIM_ * DIM_ * 2);
  bf16*  qpre  = (bf16*)take(T * DIM_ * 2);
  bf16*  qb    = (bf16*)take(T * DIM_ * 2);
  float* kf32  = (float*)take(T * (NKV_ * HD_) * 4);
  float* vf32  = (float*)take(T * (NKV_ * HD_) * 4);
  bf16*  kb    = (bf16*)take(T * (NKV_ * HD_) * 2);
  bf16*  vb    = (bf16*)take(T * (NKV_ * HD_) * 2);
  bf16*  attnb = (bf16*)take(T * DIM_ * 2);

  // 1) casts to bf16
  cast_kernel<<<dim3((unsigned)((T * DIM_) / 256)), 256, 0, stream>>>(
      x, xb, (int)(T * DIM_));
  cast_kernel<<<dim3((DIM_ * DIM_) / 256), 256, 0, stream>>>(wq, wqb,
                                                             DIM_ * DIM_);
  cast_kernel<<<dim3((NKV_ * HD_ * DIM_) / 256), 256, 0, stream>>>(
      wk, wkb, NKV_ * HD_ * DIM_);
  cast_kernel<<<dim3((NKV_ * HD_ * DIM_) / 256), 256, 0, stream>>>(
      wv, wvb, NKV_ * HD_ * DIM_);
  cast_kernel<<<dim3((DIM_ * DIM_) / 256), 256, 0, stream>>>(wo, wob,
                                                             DIM_ * DIM_);

  // 2) projections (WMMA GEMMs)
  gemm_bf16_kernel<true><<<dim3(DIM_ / 128, (unsigned)(T / 128)), 256, 0,
                           stream>>>(xb, wqb, qpre, (int)T, DIM_, DIM_);
  gemm_bf16_kernel<false><<<dim3((NKV_ * HD_) / 128, (unsigned)(T / 128)), 256,
                            0, stream>>>(xb, wkb, kf32, (int)T, NKV_ * HD_,
                                         DIM_);
  gemm_bf16_kernel<false><<<dim3((NKV_ * HD_) / 128, (unsigned)(T / 128)), 256,
                            0, stream>>>(xb, wvb, vf32, (int)T, NKV_ * HD_,
                                         DIM_);

  // 3) RoPE + K/V expansion (keys/values f32 outputs)
  rope_q_kernel<<<dim3((unsigned)((T * NH_ * 64) / 256)), 256, 0, stream>>>(
      qpre, fc, fs, qb);
  rope_k_kernel<<<dim3((unsigned)((T * NKV_ * 64) / 256)), 256, 0, stream>>>(
      kf32, fc, fs, kb, keys);
  expand_v_kernel<<<dim3((unsigned)((T * NKV_ * HD_) / 256)), 256, 0, stream>>>(
      vf32, vb, values);

  // 4) causal flash attention (WMMA)
  attn_kernel<<<dim3(B_ * NH_, L_ / 128), 256, 0, stream>>>(qb, kb, vb, attnb);

  // 5) output projection (WMMA GEMM, f32 out)
  gemm_bf16_kernel<false><<<dim3(DIM_ / 128, (unsigned)(T / 128)), 256, 0,
                            stream>>>(attnb, wob, out, (int)T, DIM_, DIM_);
}
